// SparseMoE_cv_70368744178379
// MI455X (gfx1250) — matrix operand order repaired and checked by reading, not compiled, not run on hardware
//
#include <hip/hip_runtime.h>
#include <hip/hip_bf16.h>

// ---------------- problem constants (match reference) ----------------
#define DD   512
#define EE   8
#define FF   2048
#define TT   2048            // BS*H*W = 2*32*32
#define TOPK 2
#define NASG (TT*TOPK)       // 4096 assignments
#define LSTRIDE TT           // max tokens per expert
#define KT1 (DD/32)          // 16 k-tiles, GEMM1
#define NT1 (FF/16)          // 128 n-tiles, GEMM1
#define KT2 (FF/32)          // 64 k-tiles, GEMM2
#define NT2 (DD/16)          // 32 n-tiles, GEMM2
#define NFRAG1 (EE*KT1*NT1)  // 16384 B-fragments for W1
#define NFRAG2 (EE*KT2*NT2)  // 16384 B-fragments for W2
#define LDH (FF + 8)         // hidden LDS row stride (halves): 4112B, bank-conflict-free

typedef __attribute__((ext_vector_type(16))) __bf16 v16bf;
typedef __attribute__((ext_vector_type(8)))  __bf16 v8bf;
typedef __attribute__((ext_vector_type(8)))  float  v8f;

// ---------------- 1) router: exact f32, one wave per token ----------------
__global__ __launch_bounds__(256) void router_kernel(
    const float* __restrict__ x, const float* __restrict__ Wg, const float* __restrict__ bg,
    const float* __restrict__ Wn, const float* __restrict__ bn,
    const float* __restrict__ noise_eps,
    int* __restrict__ expert_of, float* __restrict__ gate_of)
{
  const int lane = threadIdx.x & 31;
  const int wave = threadIdx.x >> 5;
  const int t = blockIdx.x * 8 + wave;
  const int b = t >> 10;
  const int hw = t & 1023;
  const float* xp = x + (size_t)b * DD * 1024 + hw;   // x[(b*D+d)*1024 + hw]

  float accg[EE], accn[EE];
#pragma unroll
  for (int e = 0; e < EE; ++e) { accg[e] = 0.f; accn[e] = 0.f; }

  for (int it = 0; it < DD / 32; ++it) {
    int d = it * 32 + lane;
    float xv = xp[(size_t)d * 1024];
    const float* wg = Wg + d * EE;
    const float* wn = Wn + d * EE;
#pragma unroll
    for (int e = 0; e < EE; ++e) {
      accg[e] += xv * wg[e];
      accn[e] += xv * wn[e];
    }
  }
  // full wave reduction (all lanes end with the total)
#pragma unroll
  for (int off = 16; off >= 1; off >>= 1) {
#pragma unroll
    for (int e = 0; e < EE; ++e) {
      accg[e] += __shfl_xor(accg[e], off, 32);
      accn[e] += __shfl_xor(accn[e], off, 32);
    }
  }
  if (lane == 0) {
    float noisy[EE];
#pragma unroll
    for (int e = 0; e < EE; ++e) {
      float lg = accg[e] + bg[e];
      float nr = accn[e] + bn[e];
      float sp = fmaxf(nr, 0.f) + log1pf(expf(-fabsf(nr)));   // stable softplus
      noisy[e] = lg + noise_eps[t * EE + e] * sp;
    }
    int i0 = 0;
#pragma unroll
    for (int e = 1; e < EE; ++e) if (noisy[e] > noisy[i0]) i0 = e;   // first max (lowest idx on tie)
    int i1 = -1;
#pragma unroll
    for (int e = 0; e < EE; ++e)
      if (e != i0 && (i1 < 0 || noisy[e] > noisy[i1])) i1 = e;
    float m  = noisy[i0];
    float e1 = expf(noisy[i1] - m);
    float s  = 1.f + e1;                 // softmax over the two kept entries
    expert_of[t * 2 + 0] = i0; gate_of[t * 2 + 0] = 1.f / s;
    expert_of[t * 2 + 1] = i1; gate_of[t * 2 + 1] = e1 / s;
  }
}

// ---------------- 2) stable counting sort into per-expert lists (1 wave) ----------------
__global__ void build_lists_kernel(const int* __restrict__ expert_of,
                                   int* __restrict__ list, int* __restrict__ counts)
{
  const int lane = threadIdx.x;
  int base[EE];
#pragma unroll
  for (int e = 0; e < EE; ++e) base[e] = 0;

  for (int c = 0; c < NASG / 32; ++c) {
    int a  = c * 32 + lane;
    int ex = expert_of[a];
#pragma unroll
    for (int e = 0; e < EE; ++e) {
      unsigned long long m = __ballot(ex == e);
      if (ex == e) {
        int pos = base[e] + __popcll(m & ((1ull << lane) - 1ull));
        list[e * LSTRIDE + pos] = a;           // a encodes (token<<1)|slot
      }
      base[e] += __popcll(m);
    }
  }
  if (lane == 0) {
#pragma unroll
    for (int e = 0; e < EE; ++e) counts[e] = base[e];
  }
}

// ---------------- 3) x[B,D,H,W] -> tokens_bf16[T,D], LDS-tiled transpose ----------------
__global__ __launch_bounds__(256) void pack_tokens_kernel(const float* __restrict__ x,
                                                          __bf16* __restrict__ tok)
{
  __shared__ float tile[32][33];
  const int tt = blockIdx.x % (TT / 32);   // token tile (64)
  const int dt = blockIdx.x / (TT / 32);   // d tile (16)
  const int tl = threadIdx.x & 31;
  const int rr = threadIdx.x >> 5;         // 0..7
  const int tbase = tt * 32;
  const int b = tbase >> 10;
  const int hwb = tbase & 1023;
#pragma unroll
  for (int p = 0; p < 4; ++p) {
    int dl = rr + p * 8;
    int d  = dt * 32 + dl;
    tile[tl][dl] = x[(size_t)(b * DD + d) * 1024 + hwb + tl];   // coalesced over tl
  }
  __syncthreads();
#pragma unroll
  for (int p = 0; p < 4; ++p) {
    int t2 = rr + p * 8;
    tok[(size_t)(tbase + t2) * DD + dt * 32 + tl] = (__bf16)tile[t2][tl]; // coalesced over tl
  }
}

// ---------------- 4) W1/W2 f32 -> bf16 WMMA-B fragment layout ----------------
// B packing used here (contiguous k, not the ISA table order): lane l -> N = l&15 ; element i -> K = 16*(l>>4) + i
__global__ __launch_bounds__(256) void pack_weights_kernel(
    const float* __restrict__ W1, const float* __restrict__ W2,
    __bf16* __restrict__ w1f, __bf16* __restrict__ w2f)
{
  const int gw   = blockIdx.x * 8 + (threadIdx.x >> 5);
  const int lane = threadIdx.x & 31;
  const int lhi  = lane >> 4, llo = lane & 15;
  v16bf v;
  if (gw < NFRAG1) {
    int nn = gw % NT1, kk = (gw / NT1) % KT1, e = gw / (NT1 * KT1);
    const float* src = W1 + ((size_t)e * DD + kk * 32 + lhi * 16) * FF + nn * 16 + llo;
#pragma unroll
    for (int i = 0; i < 16; ++i) v[i] = (__bf16)src[(size_t)i * FF];
    *(v16bf*)(w1f + (size_t)gw * 512 + lane * 16) = v;
  } else {
    int g2 = gw - NFRAG1;
    int nn = g2 % NT2, kk = (g2 / NT2) % KT2, e = g2 / (NT2 * KT2);
    const float* src = W2 + ((size_t)e * FF + kk * 32 + lhi * 16) * DD + nn * 16 + llo;
#pragma unroll
    for (int i = 0; i < 16; ++i) v[i] = (__bf16)src[(size_t)i * DD];
    *(v16bf*)(w2f + (size_t)g2 * 512 + lane * 16) = v;
  }
}

// ---------------- 5) fused expert MLP: 16 tokens x one expert per block ----------------
// A order used here (contiguous k, same as B packing): lane l -> M = l&15 ; element i -> K = 16*(l>>4) + i
__global__ __launch_bounds__(256) void moe_gemm_kernel(
    const __bf16* __restrict__ tok, const __bf16* __restrict__ w1f, const __bf16* __restrict__ w2f,
    const float* __restrict__ b1, const float* __restrict__ b2,
    const int* __restrict__ list, const int* __restrict__ counts,
    float* __restrict__ pair)
{
  __shared__ __align__(32) __bf16 hidden[16 * LDH];   // 16 x (2048+8) bf16 = 64.3 KB
  __shared__ int asg[16];
  __shared__ int tokid[16];

  const int e    = blockIdx.y;
  const int tile = blockIdx.x;
  const int cnt  = counts[e];
  if (tile * 16 >= cnt) return;                 // uniform early-exit

  if (threadIdx.x < 16) {
    int idx = tile * 16 + threadIdx.x;
    int a   = (idx < cnt) ? list[e * LSTRIDE + idx] : -1;
    asg[threadIdx.x]   = a;
    tokid[threadIdx.x] = (a >= 0) ? (a >> 1) : 0;
  }
  __syncthreads();

  const int lane = threadIdx.x & 31;
  const int w    = threadIdx.x >> 5;            // 8 waves
  const int lhi  = lane >> 4, llo = lane & 15;
  const __bf16* arow = tok + (size_t)tokid[llo] * DD;

  // ---- phase 1: hidden = relu(tokens @ W1e + b1e), bf16 into LDS ----
  for (int j = 0; j < 16; ++j) {                // each wave: 16 N-tiles (256 f-cols)
    const int nn = w * 16 + j;
    v8f c = {0.f,0.f,0.f,0.f,0.f,0.f,0.f,0.f};
#pragma unroll 4
    for (int kk = 0; kk < KT1; ++kk) {
      v8bf alo = *(const v8bf*)(arow + kk * 32 + lhi * 16);
      v8bf ahi = *(const v8bf*)(arow + kk * 32 + lhi * 16 + 8);
      v16bf a  = __builtin_shufflevector(alo, ahi, 0,1,2,3,4,5,6,7,8,9,10,11,12,13,14,15);
      v16bf bb = *(const v16bf*)(w1f + (((size_t)e * KT1 + kk) * NT1 + nn) * 512 + lane * 16);
      c = __builtin_amdgcn_wmma_f32_16x16x32_bf16(false, a, false, bb, (short)0, c, false, false);
    }
    const int f = nn * 16 + llo;
    const float bv = b1[e * FF + f];
#pragma unroll
    for (int r = 0; r < 8; ++r) {
      float vv = fmaxf(c[r] + bv, 0.f);
      hidden[(r + 8 * lhi) * LDH + f] = (__bf16)vv;   // C layout: M=r+8*lhi, N=llo
    }
  }
  __syncthreads();

  // ---- phase 2: out_e = hidden @ W2e + b2e, scatter to pair buffer ----
  v8f acc[4];
#pragma unroll
  for (int j = 0; j < 4; ++j) acc[j] = (v8f){0.f,0.f,0.f,0.f,0.f,0.f,0.f,0.f};

  for (int kk = 0; kk < KT2; ++kk) {
    const __bf16* hr = hidden + llo * LDH + kk * 32 + lhi * 16;  // conflict-free b128 reads
    v8bf alo = *(const v8bf*)(hr);
    v8bf ahi = *(const v8bf*)(hr + 8);
    v16bf a  = __builtin_shufflevector(alo, ahi, 0,1,2,3,4,5,6,7,8,9,10,11,12,13,14,15);
#pragma unroll
    for (int j = 0; j < 4; ++j) {               // each wave: 4 N-tiles (64 d-cols)
      const int nn = w * 4 + j;
      v16bf bb = *(const v16bf*)(w2f + (((size_t)e * KT2 + kk) * NT2 + nn) * 512 + lane * 16);
      acc[j] = __builtin_amdgcn_wmma_f32_16x16x32_bf16(false, a, false, bb, (short)0, acc[j], false, false);
    }
  }
#pragma unroll
  for (int j = 0; j < 4; ++j) {
    const int n   = (w * 4 + j) * 16 + llo;
    const float bv = b2[e * DD + n];
#pragma unroll
    for (int r = 0; r < 8; ++r) {
      int m = r + 8 * lhi;
      int a = asg[m];
      if (a >= 0) pair[(size_t)a * DD + n] = acc[j][r] + bv;   // one writer per (t,slot)
    }
  }
}

// ---------------- 6) combine: out[b,d,h,w] = g0*pair[2t] + g1*pair[2t+1] ----------------
__global__ __launch_bounds__(256) void combine_kernel(
    const float* __restrict__ pair, const float* __restrict__ gate_of, float* __restrict__ out)
{
  const int idx = blockIdx.x * 256 + threadIdx.x;   // T*D threads
  const int t = idx & (TT - 1);
  const int d = idx >> 11;                          // idx / 2048
  const float g0 = gate_of[2 * t], g1 = gate_of[2 * t + 1];
  const float v = g0 * pair[(size_t)(2 * t) * DD + d] +
                  g1 * pair[(size_t)(2 * t + 1) * DD + d];
  const int b = t >> 10, hw = t & 1023;
  out[(size_t)(b * DD + d) * 1024 + hw] = v;        // coalesced over hw
}

// ---------------- host: workspace layout + launches ----------------
static inline size_t align256(size_t x) { return (x + 255) & ~(size_t)255; }

extern "C" void kernel_launch(void* const* d_in, const int* in_sizes, int n_in,
                              void* d_out, int out_size, void* d_ws, size_t ws_size,
                              hipStream_t stream) {
  (void)in_sizes; (void)n_in; (void)out_size; (void)ws_size;
  const float* x   = (const float*)d_in[0];
  const float* Wg  = (const float*)d_in[1];
  const float* bg  = (const float*)d_in[2];
  const float* Wn  = (const float*)d_in[3];
  const float* bn  = (const float*)d_in[4];
  const float* W1  = (const float*)d_in[5];
  const float* b1  = (const float*)d_in[6];
  const float* W2  = (const float*)d_in[7];
  const float* b2  = (const float*)d_in[8];
  const float* eps = (const float*)d_in[9];
  // d_in[10] = top_k (==2, hardcoded)

  char* ws = (char*)d_ws;
  size_t off = 0;
  __bf16* tokb = (__bf16*)(ws + off); off = align256(off + (size_t)TT * DD * 2);
  __bf16* w1f  = (__bf16*)(ws + off); off = align256(off + (size_t)NFRAG1 * 512 * 2);
  __bf16* w2f  = (__bf16*)(ws + off); off = align256(off + (size_t)NFRAG2 * 512 * 2);
  float*  pair = (float*)(ws + off);  off = align256(off + (size_t)NASG * DD * 4);
  int*    eof  = (int*)(ws + off);    off = align256(off + (size_t)NASG * 4);
  float*  gof  = (float*)(ws + off);  off = align256(off + (size_t)NASG * 4);
  int*    list = (int*)(ws + off);    off = align256(off + (size_t)EE * LSTRIDE * 4);
  int*    cnts = (int*)(ws + off);    off = align256(off + 256);

  router_kernel<<<TT / 8, 256, 0, stream>>>(x, Wg, bg, Wn, bn, eps, eof, gof);
  build_lists_kernel<<<1, 32, 0, stream>>>(eof, list, cnts);
  pack_tokens_kernel<<<(TT / 32) * (DD / 32), 256, 0, stream>>>(x, tokb);
  pack_weights_kernel<<<(NFRAG1 + NFRAG2) / 8, 256, 0, stream>>>(W1, W2, w1f, w2f);
  moe_gemm_kernel<<<dim3(TT / 16, EE), 256, 0, stream>>>(tokb, w1f, w2f, b1, b2,
                                                         list, cnts, pair);
  combine_kernel<<<(TT * DD) / 256, 256, 0, stream>>>(pair, gof, (float*)d_out);
}
